// Decoder_25400436589038
// MI455X (gfx1250) — matrix operand order repaired and checked by reading, not compiled, not run on hardware
//
#include <hip/hip_runtime.h>
#include <stdint.h>
#include <math.h>

#define VOCAB  32000
#define EMBED  512
#define HIDDEN 1024
#define BATCH  128
#define MAXLEN 48
#define GATES  (4 * HIDDEN)        // 4096
#define MTILES (BATCH / 16)        // 8

typedef __attribute__((ext_vector_type(16))) __bf16 v16bf;
typedef __attribute__((ext_vector_type(8)))  __bf16 bf16x8;
typedef __attribute__((ext_vector_type(8)))  float  v8f;

union Frag { v16bf v; bf16x8 h[2]; };

static __device__ inline bf16x8 ld8(const __bf16* p) {
    return *reinterpret_cast<const bf16x8*>(p);
}

static __device__ inline uint16_t f2bf(float f) {
    unsigned u = __float_as_uint(f);
    unsigned r = 0x7FFFu + ((u >> 16) & 1u);   // round-to-nearest-even
    return (uint16_t)((u + r) >> 16);
}

static __device__ inline float sigm(float x) {
    return 1.0f / (1.0f + __expf(-x));
}

// ---------------- setup kernels ----------------

__global__ void k_f32_to_bf16(const float* __restrict__ src,
                              uint16_t* __restrict__ dst, int n) {
    int i = blockIdx.x * blockDim.x + threadIdx.x;
    if (i < n) dst[i] = f2bf(src[i]);
}

__global__ void k_init(const float* __restrict__ eh, const float* __restrict__ ec,
                       float* __restrict__ c, uint16_t* __restrict__ hbf,
                       float* __restrict__ bias,
                       const float* __restrict__ bih, const float* __restrict__ bhh,
                       int* __restrict__ tok, const int* __restrict__ sos) {
    int i = blockIdx.x * blockDim.x + threadIdx.x;
    if (i < BATCH * HIDDEN) {
        c[i]   = ec[i];
        hbf[i] = f2bf(eh[i]);
    }
    if (i < GATES) bias[i] = bih[i] + bhh[i];
    if (i < BATCH) tok[i]  = sos[0];
}

__global__ void k_zero(float* __restrict__ p, int n) {
    int i = blockIdx.x * blockDim.x + threadIdx.x;
    if (i < n) p[i] = 0.0f;
}

// ---------------- gates GEMM: [x|h] @ [w_ih|w_hh]^T + bias ----------------
// grid = GATES/64 blocks of 128 threads (4 waves).
// Each wave owns a 16-wide N strip and ALL of M=128 via 8 accumulators,
// so each weight fragment is loaded exactly once per step.

__global__ __launch_bounds__(128)
void k_gates(const uint16_t* __restrict__ embq, const uint16_t* __restrict__ wihq,
             const uint16_t* __restrict__ whhq, const uint16_t* __restrict__ hbfq,
             const int* __restrict__ tok, const float* __restrict__ bias,
             float* __restrict__ gates) {
    const __bf16* emb = (const __bf16*)embq;
    const __bf16* wih = (const __bf16*)wihq;
    const __bf16* whh = (const __bf16*)whhq;
    const __bf16* hbf = (const __bf16*)hbfq;

    const int lane = threadIdx.x & 31;
    const int wave = threadIdx.x >> 5;
    const int r    = lane & 15;
    const int hi   = lane >> 4;

    const int nB = blockIdx.x * 64 + wave * 16 + r;   // this lane's B column
    const __bf16* wihRow = wih + (size_t)nB * EMBED;
    const __bf16* whhRow = whh + (size_t)nB * HIDDEN;

    v8f acc[MTILES];
#pragma unroll
    for (int m = 0; m < MTILES; ++m) acc[m] = (v8f){};

    // ---- K region 1: x (gathered embedding rows) x w_ih, K = 0..511 ----
    {
        const __bf16* aRow[MTILES];
#pragma unroll
        for (int m = 0; m < MTILES; ++m)
            aRow[m] = emb + (size_t)tok[m * 16 + r] * EMBED;

#pragma unroll 2
        for (int k0 = 0; k0 < EMBED; k0 += 32) {
            Frag b;
            b.h[0] = ld8(wihRow + k0 + hi * 8);
            b.h[1] = ld8(wihRow + k0 + 16 + hi * 8);
#pragma unroll
            for (int m = 0; m < MTILES; ++m) {
                Frag a;
                a.h[0] = ld8(aRow[m] + k0 + hi * 8);
                a.h[1] = ld8(aRow[m] + k0 + 16 + hi * 8);
                acc[m] = __builtin_amdgcn_wmma_f32_16x16x32_bf16(
                    false, a.v, false, b.v, (short)0, acc[m], false, false);
            }
        }
    }

    // ---- K region 2: h x w_hh, K = 0..1023 ----
    {
        const __bf16* aRow[MTILES];
#pragma unroll
        for (int m = 0; m < MTILES; ++m)
            aRow[m] = hbf + (size_t)(m * 16 + r) * HIDDEN;

#pragma unroll 2
        for (int k0 = 0; k0 < HIDDEN; k0 += 32) {
            Frag b;
            b.h[0] = ld8(whhRow + k0 + hi * 8);
            b.h[1] = ld8(whhRow + k0 + 16 + hi * 8);
#pragma unroll
            for (int m = 0; m < MTILES; ++m) {
                Frag a;
                a.h[0] = ld8(aRow[m] + k0 + hi * 8);
                a.h[1] = ld8(aRow[m] + k0 + 16 + hi * 8);
                acc[m] = __builtin_amdgcn_wmma_f32_16x16x32_bf16(
                    false, a.v, false, b.v, (short)0, acc[m], false, false);
            }
        }
    }

    const float bn = bias[nB];
#pragma unroll
    for (int m = 0; m < MTILES; ++m) {
#pragma unroll
        for (int j = 0; j < 8; ++j) {
            int row = m * 16 + j + hi * 8;   // D layout: VGPR j -> M=j (lanes<16), M=j+8 (lanes>=16)
            gates[(size_t)row * GATES + nB] = acc[m][j] + bn;
        }
    }
}

// ---------------- LSTM pointwise ----------------

__global__ void k_lstm(const float* __restrict__ gates,
                       float* __restrict__ c, uint16_t* __restrict__ hbf) {
    int i = blockIdx.x * blockDim.x + threadIdx.x;
    if (i >= BATCH * HIDDEN) return;
    int m = i >> 10, j = i & (HIDDEN - 1);
    const float* g = gates + (size_t)m * GATES;
    float ig = sigm(g[j]);
    float fg = sigm(g[j + HIDDEN]);
    float gg = tanhf(g[j + 2 * HIDDEN]);
    float og = sigm(g[j + 3 * HIDDEN]);
    float cn = fg * c[i] + ig * gg;
    float hn = og * tanhf(cn);
    c[i]   = cn;
    hbf[i] = f2bf(hn);
}

// ---------------- logits GEMM: h @ w_out^T + b_out ----------------
// grid = VOCAB/64 blocks of 128 threads; each wave: 16-wide N strip, all of M.
// w_out streams through L2 exactly once per step.

__global__ __launch_bounds__(128)
void k_logits(const uint16_t* __restrict__ hbfq, const uint16_t* __restrict__ woutq,
              const float* __restrict__ bout, float* __restrict__ out) {
    const __bf16* hbf  = (const __bf16*)hbfq;
    const __bf16* wout = (const __bf16*)woutq;

    const int lane = threadIdx.x & 31;
    const int wave = threadIdx.x >> 5;
    const int r    = lane & 15;
    const int hi   = lane >> 4;

    const int nB = blockIdx.x * 64 + wave * 16 + r;
    const __bf16* bRow = wout + (size_t)nB * HIDDEN;

    const __bf16* aRow[MTILES];
#pragma unroll
    for (int m = 0; m < MTILES; ++m)
        aRow[m] = hbf + (size_t)(m * 16 + r) * HIDDEN;

    v8f acc[MTILES];
#pragma unroll
    for (int m = 0; m < MTILES; ++m) acc[m] = (v8f){};

#pragma unroll 2
    for (int k0 = 0; k0 < HIDDEN; k0 += 32) {
        Frag b;
        b.h[0] = ld8(bRow + k0 + hi * 8);
        b.h[1] = ld8(bRow + k0 + 16 + hi * 8);
#pragma unroll
        for (int m = 0; m < MTILES; ++m) {
            Frag a;
            a.h[0] = ld8(aRow[m] + k0 + hi * 8);
            a.h[1] = ld8(aRow[m] + k0 + 16 + hi * 8);
            acc[m] = __builtin_amdgcn_wmma_f32_16x16x32_bf16(
                false, a.v, false, b.v, (short)0, acc[m], false, false);
        }
    }

    const float bn = bout[nB];
#pragma unroll
    for (int m = 0; m < MTILES; ++m) {
#pragma unroll
        for (int j = 0; j < 8; ++j) {
            int row = m * 16 + j + hi * 8;
            out[(size_t)row * VOCAB + nB] = acc[m][j] + bn;
        }
    }
}

// ---------------- greedy argmax: one block per batch row ----------------

__global__ __launch_bounds__(256)
void k_argmax(const float* __restrict__ logits, int* __restrict__ tok) {
    const int m = blockIdx.x;
    const float* row = logits + (size_t)m * VOCAB;
    float best = -INFINITY;
    int   bi   = 0;
    for (int cidx = threadIdx.x; cidx < VOCAB; cidx += 256) {
        float v = row[cidx];
        if (v > best) { best = v; bi = cidx; }   // keeps first max (ascending scan)
    }
    __shared__ float sv[256];
    __shared__ int   si[256];
    sv[threadIdx.x] = best;
    si[threadIdx.x] = bi;
    __syncthreads();
    for (int s = 128; s > 0; s >>= 1) {
        if (threadIdx.x < s) {
            float v2 = sv[threadIdx.x + s];
            int   i2 = si[threadIdx.x + s];
            if (v2 > sv[threadIdx.x] ||
                (v2 == sv[threadIdx.x] && i2 < si[threadIdx.x])) {
                sv[threadIdx.x] = v2;
                si[threadIdx.x] = i2;
            }
        }
        __syncthreads();
    }
    if (threadIdx.x == 0) tok[m] = si[0];
}

// ---------------- host ----------------

extern "C" void kernel_launch(void* const* d_in, const int* in_sizes, int n_in,
                              void* d_out, int out_size, void* d_ws, size_t ws_size,
                              hipStream_t stream) {
    const float* encoder_h = (const float*)d_in[0];
    const float* encoder_c = (const float*)d_in[1];
    const float* embedding = (const float*)d_in[2];
    const float* w_ih      = (const float*)d_in[3];
    const float* w_hh      = (const float*)d_in[4];
    const float* b_ih      = (const float*)d_in[5];
    const float* b_hh      = (const float*)d_in[6];
    const float* w_out     = (const float*)d_in[7];
    const float* b_out     = (const float*)d_in[8];
    const int*   sos_id    = (const int*)d_in[9];
    float* out = (float*)d_out;

    // workspace layout (256B aligned slices)
    char* base = (char*)d_ws;
    size_t off = 0;
    auto take = [&](size_t bytes) -> char* {
        char* p = base + off;
        off = (off + bytes + 255) & ~(size_t)255;
        return p;
    };
    uint16_t* emb_bf  = (uint16_t*)take((size_t)VOCAB * EMBED  * 2);
    uint16_t* wih_bf  = (uint16_t*)take((size_t)GATES * EMBED  * 2);
    uint16_t* whh_bf  = (uint16_t*)take((size_t)GATES * HIDDEN * 2);
    uint16_t* wout_bf = (uint16_t*)take((size_t)VOCAB * HIDDEN * 2);
    float*    bias    = (float*)   take((size_t)GATES * 4);
    float*    c_st    = (float*)   take((size_t)BATCH * HIDDEN * 4);
    uint16_t* h_bf    = (uint16_t*)take((size_t)BATCH * HIDDEN * 2);
    float*    gates   = (float*)   take((size_t)BATCH * GATES * 4);
    int*      tok     = (int*)     take((size_t)BATCH * 4);
    (void)ws_size; (void)in_sizes; (void)n_in; (void)out_size;

    // per-launch weight conversion f32 -> bf16 : halves L2-bound weight traffic
    {
        int n;
        n = VOCAB * EMBED;
        k_f32_to_bf16<<<(n + 255) / 256, 256, 0, stream>>>(embedding, emb_bf, n);
        n = GATES * EMBED;
        k_f32_to_bf16<<<(n + 255) / 256, 256, 0, stream>>>(w_ih, wih_bf, n);
        n = GATES * HIDDEN;
        k_f32_to_bf16<<<(n + 255) / 256, 256, 0, stream>>>(w_hh, whh_bf, n);
        n = VOCAB * HIDDEN;
        k_f32_to_bf16<<<(n + 255) / 256, 256, 0, stream>>>(w_out, wout_bf, n);
    }
    k_init<<<(BATCH * HIDDEN + 255) / 256, 256, 0, stream>>>(
        encoder_h, encoder_c, c_st, h_bf, bias, b_ih, b_hh, tok, sos_id);
    // outputs[0] stays zeros
    k_zero<<<(BATCH * VOCAB + 255) / 256, 256, 0, stream>>>(out, BATCH * VOCAB);

    for (int t = 1; t < MAXLEN; ++t) {
        k_gates<<<GATES / 64, 128, 0, stream>>>(
            emb_bf, wih_bf, whh_bf, h_bf, tok, bias, gates);
        k_lstm<<<(BATCH * HIDDEN + 255) / 256, 256, 0, stream>>>(gates, c_st, h_bf);
        float* plane = out + (size_t)t * BATCH * VOCAB;
        k_logits<<<VOCAB / 64, 128, 0, stream>>>(h_bf, wout_bf, b_out, plane);
        if (t + 1 < MAXLEN) {
            k_argmax<<<BATCH, 256, 0, stream>>>(plane, tok);
        }
    }
}
